// _DynamicSnakeConvBase_28905129902138
// MI455X (gfx1250) — matrix operand order repaired and checked by reading, not compiled, not run on hardware
//
#include <hip/hip_runtime.h>

typedef __attribute__((ext_vector_type(16))) _Float16 v16h;
typedef __attribute__((ext_vector_type(8)))  float    v8f;

static constexpr int Bn   = 4;
static constexpr int Cin  = 64;
static constexpr int CoutN= 64;
static constexpr int Hh   = 160;
static constexpr int Ww   = 160;
static constexpr int Kk   = 9;
static constexpr int PADh = 4;
static constexpr int HW   = Hh * Ww;        // 25600
static constexpr int NPOS = Bn * HW;        // 102400
static constexpr int RDIM = Kk * Cin;       // 576
static constexpr int NSTEP= RDIM / 32;      // 18 K-steps of 32

// ---------------------------------------------------------------------------
// Kernel 0: zero BN stat accumulators; build two pre-swizzled f16 B-fragment
// buffers:
//   wsw    [NSTEP][4 nt][32 lane][16 e]  <- w_def  (deform GEMM, N=64)
//   woffsw [NSTEP][1 nt][32 lane][16 e]  <- w_off odd channels (N=16, 9 used)
// B-fragment: K_local = (e&7) + 8*(lane>>4) + 16*(e>>3), N = nt*16 + (lane&15).
// ---------------------------------------------------------------------------
__global__ void k_prep(const float* __restrict__ w_def,
                       const float* __restrict__ w_off,
                       float* __restrict__ stats,
                       _Float16* __restrict__ wsw,
                       _Float16* __restrict__ woffsw) {
    int t = blockIdx.x * 256 + threadIdx.x;
    if (blockIdx.x == 0 && threadIdx.x < 18) stats[threadIdx.x] = 0.0f;
    if (t < RDIM * CoutN) {
        int e  = t & 15;
        int l  = (t >> 4) & 31;
        int nt = (t >> 9) & 3;
        int s  = t >> 11;
        int g  = l >> 4;
        int n  = nt * 16 + (l & 15);
        int kg = s * 32 + (e & 7) + 8 * g + 16 * (e >> 3); // r = kk*64 + c
        int c  = kg & 63;
        int kk = kg >> 6;
        wsw[t] = (_Float16)w_def[(n * Cin + c) * Kk + kk];
    } else if (t - RDIM * CoutN < NSTEP * 32 * 16) {
        int t2 = t - RDIM * CoutN;
        int e  = t2 & 15;
        int l  = (t2 >> 4) & 31;
        int s  = t2 >> 9;
        int g  = l >> 4;
        int j  = l & 15;
        int kg = s * 32 + (e & 7) + 8 * g + 16 * (e >> 3); // r = ky*64 + c
        int c  = kg & 63;
        int ky = kg >> 6;
        woffsw[t2] = (j < 9) ? (_Float16)w_off[(2 * j + 1) * 576 + c * 9 + ky]
                             : (_Float16)0.0f;
    }
}

// ---------------------------------------------------------------------------
// Kernel 1 (WMMA): 9x1 vertical conv, odd offset channels only, as a GEMM:
// M = 128 positions/block (8 waves x 16x16 tile), N = 16 (9 used), R = 576.
// Double-buffered per-step staging (one barrier/step), branch-free zero-pad
// via clamped row + float mask, all per-thread indices in registers.
// ---------------------------------------------------------------------------
__global__ void __launch_bounds__(256)
k_offconv(const float* __restrict__ x, const float* __restrict__ b_off,
          const _Float16* __restrict__ woffsw, float* __restrict__ off_raw,
          float* __restrict__ stats) {
    __shared__ __align__(32) _Float16 Ald[2 * 8 * 32 * 16]; // 16 KB, 2 buffers
    __shared__ float sacc[18];

    int tid = threadIdx.x;
    int posBase = blockIdx.x * 128;            // HW%128==0 -> single b per block
    int b = posBase / HW;
    int remBase = posBase - b * HW;

    if (tid < 18) sacc[tid] = 0.0f;

    // Per-thread staging indices: position p = tid&127 is FIXED.
    int p   = tid & 127;
    int rp0 = tid >> 7;                        // 0 or 1
    int rem = remBase + p;
    int h   = rem / Ww;
    int w   = rem - h * Ww;
    int mtS = p >> 4, mS = p & 15;
    const float* xw = x + (size_t)b * Cin * HW + w;

    auto stage = [&](int s, int bufsel) {
#pragma unroll
        for (int it = 0; it < 8; ++it) {
            int kl = 2 * (rp0 + 2 * it);       // even K_local 0..30
            int r  = s * 32 + kl;
            int c  = r & 63;                   // even; c+1 has same ky
            int ky = r >> 6;
            int row  = h - PADh + ky;
            int rowc = min(max(row, 0), Hh - 1);
            float msk = ((unsigned)row < (unsigned)Hh) ? 1.0f : 0.0f;
            const float* xp = xw + (size_t)c * HW + rowc * Ww;
            float v0 = xp[0]  * msk;
            float v1 = xp[HW] * msk;
            int g = (kl >> 3) & 1;
            int e = (kl & 7) + 8 * (kl >> 4);  // even
            union { _Float16 h2[2]; unsigned u; } pk;
            pk.h2[0] = (_Float16)v0;
            pk.h2[1] = (_Float16)v1;
            *(unsigned*)&Ald[bufsel * 4096 + ((mtS * 32 + (g * 16 + mS)) * 16 + e)] = pk.u;
        }
    };

    int wv = tid >> 5, lane = tid & 31;
    v8f acc = {};
    stage(0, 0);
    for (int s = 0; s < NSTEP; ++s) {
        __syncthreads();                       // staging of step s visible
        v16h a  = *(v16h*)&Ald[(s & 1) * 4096 + (wv * 32 + lane) * 16];
        v16h bb = *(const v16h*)&woffsw[(s * 32 + lane) * 16];
        if (s + 1 < NSTEP) stage(s + 1, (s + 1) & 1);
        acc = __builtin_amdgcn_wmma_f32_16x16x32_f16(
            false, a, false, bb, (short)0, acc, false, false);
    }

    // D: VGPR i -> M = i + 8*g (position within 16-tile), N = lane&15 = j
    int j = lane & 15;
    int g = lane >> 4;
    if (j < 9) {
        float bias = b_off[2 * j + 1];
        float* op = off_raw + (size_t)(b * 9 + j) * HW + remBase + wv * 16 + 8 * g;
        float sum = 0.0f, sq = 0.0f;
#pragma unroll
        for (int i = 0; i < 8; ++i) {
            float v = acc[i] + bias;
            op[i] = v;
            sum += v;
            sq  += v * v;
        }
        atomicAdd(&sacc[j], sum);
        atomicAdd(&sacc[9 + j], sq);
    }
    __syncthreads();
    if (tid < 18) atomicAdd(&stats[tid], sacc[tid]);
}

// ---------------------------------------------------------------------------
// Kernel 2: finalize BN -> per-channel scale/shift (population variance).
// ---------------------------------------------------------------------------
__global__ void k_bnstats(const float* __restrict__ gamma,
                          const float* __restrict__ beta,
                          float* __restrict__ stats) {
    int j = threadIdx.x;
    if (j < 9) {
        const float n = (float)NPOS;
        float mean = stats[j] / n;
        float var  = stats[9 + j] / n - mean * mean;
        float rs   = rsqrtf(var + 1e-5f);
        float sc   = gamma[2 * j + 1] * rs;
        stats[18 + j] = sc;
        stats[27 + j] = beta[2 * j + 1] - mean * sc;
    }
}

// ---------------------------------------------------------------------------
// Kernel 3: normalize + tanh + offset conversion.
// dx[j] = sum_{t=j..4} y[t] (j<4)  /  sum_{t=4..j} y[t] (j>=4);  dy == 0.
// ---------------------------------------------------------------------------
__global__ void __launch_bounds__(256)
k_offsets(const float* __restrict__ off_raw, const float* __restrict__ stats,
          float* __restrict__ dxbuf) {
    int gid = blockIdx.x * 256 + threadIdx.x;
    int b = gid / HW, rem = gid - b * HW;
    float y[9];
#pragma unroll
    for (int j = 0; j < 9; ++j) {
        float v = off_raw[((size_t)b * 9 + j) * HW + rem];
        y[j] = tanhf(fmaf(v, stats[18 + j], stats[27 + j]));
    }
    float dx[9];
    dx[4] = y[4];
#pragma unroll
    for (int i = 3; i >= 0; --i) dx[i] = y[i] + dx[i + 1];
#pragma unroll
    for (int i = 5; i < 9; ++i)  dx[i] = y[i] + dx[i - 1];
#pragma unroll
    for (int j = 0; j < 9; ++j)
        dxbuf[((size_t)b * 9 + j) * HW + rem] = dx[j];
}

// ---------------------------------------------------------------------------
// Kernel 4: deformable sampling + WMMA GEMM.
// Tile = 32 positions of one (b,h) row. dy==0 -> tap k reads row h-4+k with
// 1-D linear interp along w. Branch-free sampling: clamped indices x float
// masks, two channels per iteration packed into one ds_store_b32 directly in
// A-fragment layout. 8 waves: (mt 0..1)x(nt 0..3), 18 wmma each, f32 accum.
// ---------------------------------------------------------------------------
__global__ void __launch_bounds__(256)
k_deform(const float* __restrict__ x, const float* __restrict__ dxbuf,
         const _Float16* __restrict__ wsw, const float* __restrict__ b_def,
         float* __restrict__ out) {
    __shared__ __align__(32) _Float16 Ald[NSTEP * 2 * 32 * 16]; // 36 KB
    __shared__ int   rowv[9];
    __shared__ int   xloL[9 * 32];
    __shared__ float fxL[9 * 32];

    int tid = threadIdx.x;
    int tix = blockIdx.x;
    int tilesPerRow = Ww / 32;                 // 5
    int b   = tix / (Hh * tilesPerRow);
    int rem = tix - b * (Hh * tilesPerRow);
    int h   = rem / tilesPerRow;
    int w0  = (rem - h * tilesPerRow) * 32;

    // warm L0/L1 for the weight-fragment stream (global_prefetch_b8)
    for (int off = tid * 128; off < RDIM * CoutN * 2; off += 256 * 128)
        __builtin_prefetch((const char*)wsw + off, 0, 1);

    if (tid < 9) {
        int row = h - PADh + tid;
        rowv[tid] = (row >= 0 && row < Hh) ? row : -1;
    }
    if (tid < 288) {
        int k = tid >> 5, p = tid & 31;
        float d  = dxbuf[((size_t)b * 9 + k) * HW + h * Ww + (w0 + p)];
        float px = (float)(w0 + p) + d;
        float fl = floorf(px);
        xloL[tid] = (int)fl;
        fxL[tid]  = px - fl;
    }
    __syncthreads();

    // Sampling: thread -> fixed position p, channel group cg (8 channels).
    {
        int p  = tid & 31;
        int cg = tid >> 5;                     // 0..7
        int mt = p >> 4, m = p & 15;
        for (int k = 0; k < 9; ++k) {
            int   row = rowv[k];
            int   xl  = xloL[k * 32 + p];
            float f   = fxL[k * 32 + p];
            int   rok = (row >= 0) ? 1 : 0;
            float m0  = (rok & (xl     >= 0 ? 1 : 0) & (xl     < Ww ? 1 : 0)) ? 1.0f : 0.0f;
            float m1  = (rok & (xl + 1 >= 0 ? 1 : 0) & (xl + 1 < Ww ? 1 : 0)) ? 1.0f : 0.0f;
            int xc0 = min(max(xl,     0), Ww - 1);
            int xc1 = min(max(xl + 1, 0), Ww - 1);
            const float* xr = x + ((size_t)(b * Cin) * Hh + (rok ? row : 0)) * Ww;
#pragma unroll
            for (int q = 0; q < 4; ++q) {
                int c = cg * 8 + 2 * q;
                const float* x0 = xr + (size_t)c * HW;
                float v00 = x0[xc0]      * m0;
                float v01 = x0[xc1]      * m1;
                float v10 = x0[HW + xc0] * m0;
                float v11 = x0[HW + xc1] * m1;
                float s0 = fmaf(f, v01 - v00, v00);   // (1-f)*v0 + f*v1
                float s1 = fmaf(f, v11 - v10, v10);
                int r  = k * 64 + c;
                int kl = r & 31;                      // even
                int st = r >> 5;
                int g  = (kl >> 3) & 1;
                int e  = (kl & 7) + 8 * (kl >> 4);    // even
                union { _Float16 h2[2]; unsigned u; } pk;
                pk.h2[0] = (_Float16)s0;
                pk.h2[1] = (_Float16)s1;
                *(unsigned*)&Ald[((st * 2 + mt) * 32 + (g * 16 + m)) * 16 + e] = pk.u;
            }
        }
    }
    __syncthreads();

    int wv = tid >> 5, lane = tid & 31;
    int nt = wv & 3,   mt = wv >> 2;
    v8f acc = {};
#pragma unroll 2
    for (int s = 0; s < NSTEP; ++s) {
        v16h a  = *(v16h*)&Ald[((s * 2 + mt) * 32 + lane) * 16];
        v16h bb = *(const v16h*)&wsw[((s * 4 + nt) * 32 + lane) * 16];
        acc = __builtin_amdgcn_wmma_f32_16x16x32_f16(
            false, a, false, bb, (short)0, acc, false, false);
    }

    // D: VGPR i -> M = i + 8*g, N = lane&15.
    int o = nt * 16 + (lane & 15);
    int g = lane >> 4;
    float bias = b_def[o];
    float* op = out + ((size_t)(b * CoutN + o) * Hh + h) * Ww + w0 + mt * 16 + 8 * g;
#pragma unroll
    for (int i = 0; i < 8; ++i) op[i] = acc[i] + bias;
}

// ---------------------------------------------------------------------------
extern "C" void kernel_launch(void* const* d_in, const int* in_sizes, int n_in,
                              void* d_out, int out_size, void* d_ws, size_t ws_size,
                              hipStream_t stream) {
    const float* x     = (const float*)d_in[0];
    const float* w_off = (const float*)d_in[1];
    const float* b_off = (const float*)d_in[2];
    const float* gamma = (const float*)d_in[3];
    const float* beta  = (const float*)d_in[4];
    const float* w_def = (const float*)d_in[5];
    const float* b_def = (const float*)d_in[6];
    float* out = (float*)d_out;

    // Workspace (float offsets): [0,921600) off_raw ; [921600,1843200) dx ;
    // [1843200,1843236) stats ; wsw f16 @1843264 (36864 halves) ;
    // woffsw f16 @1861696 (9216 halves).  Total ~7.5 MB.
    float* ws_f      = (float*)d_ws;
    float* off_raw   = ws_f;
    float* dxbuf     = ws_f + 921600;
    float* stats     = ws_f + 1843200;
    _Float16* wsw    = (_Float16*)(ws_f + 1843264);
    _Float16* woffsw = (_Float16*)(ws_f + 1861696);

    k_prep   <<<180,                 256, 0, stream>>>(w_def, w_off, stats, wsw, woffsw);
    k_offconv<<<NPOS / 128,          256, 0, stream>>>(x, b_off, woffsw, off_raw, stats);
    k_bnstats<<<1,                    32, 0, stream>>>(gamma, beta, stats);
    k_offsets<<<NPOS / 256,          256, 0, stream>>>(off_raw, stats, dxbuf);
    k_deform <<<Bn * Hh * (Ww / 32), 256, 0, stream>>>(x, dxbuf, wsw, b_def, out);
}
